// Eq2to2_59072980189753
// MI455X (gfx1250) — matrix operand order repaired and checked by reading, not compiled, not run on hardware
//
#include <hip/hip_runtime.h>
#include <hip/hip_bf16.h>

typedef _Float16 half_t;
typedef __attribute__((ext_vector_type(16))) _Float16 v16h;
typedef __attribute__((ext_vector_type(8)))  _Float16 v8h;
typedef __attribute__((ext_vector_type(8)))  float    v8f;

#define NN 4
#define PP 128
#define HH 64
#define UU 64
#define AVGP 49.0f

#define CAT16(lo,hi) __builtin_shufflevector((lo),(hi),0,1,2,3,4,5,6,7,8,9,10,11,12,13,14,15)

__device__ inline v16h load_frag(const half_t* p0, const half_t* p1) {
  v8h lo = *(const v8h*)p0;
  v8h hi = *(const v8h*)p1;
  return CAT16(lo, hi);
}

// ---------------- K1: row sums (over j) + diagonal, per (n,i) ----------------
__global__ void k_rowsum_diag(const float* __restrict__ x,
                              float* __restrict__ sr, float* __restrict__ dg) {
  int nb = blockIdx.x;            // n*128 + i
  int i  = nb & (PP - 1);
  int h  = threadIdx.x;           // 0..63
  const float* row = x + ((size_t)nb * PP) * HH + h;
  float s = 0.f;
  #pragma unroll 4
  for (int j = 0; j < PP; ++j) s += row[(size_t)j * HH];
  sr[nb * HH + h] = s;
  dg[nb * HH + h] = row[(size_t)i * HH];
}

// ---------------- K2: column sums (over i), per (n,j) ----------------
__global__ void k_colsum(const float* __restrict__ x, float* __restrict__ sc) {
  int nb = blockIdx.x;            // n*128 + j
  int n = nb >> 7, j = nb & (PP - 1);
  int h = threadIdx.x;
  const float* col = x + (((size_t)n * PP * PP) + j) * HH + h;
  float s = 0.f;
  #pragma unroll 4
  for (int i = 0; i < PP; ++i) s += col[(size_t)i * PP * HH];
  sc[nb * HH + h] = s;
}

// ---------------- K3: per-(n,h) scalars: sum_diag/np, sum_all/np^2 ----------------
__global__ void k_scalars(const float* __restrict__ sr, const float* __restrict__ dg,
                          const float* __restrict__ npart,
                          float* __restrict__ sd, float* __restrict__ sa) {
  int n = blockIdx.x, h = threadIdx.x;
  float np = npart[n];
  float s1 = 0.f, s2 = 0.f;
  for (int i = 0; i < PP; ++i) {
    s1 += dg[(n * PP + i) * HH + h];
    s2 += sr[(n * PP + i) * HH + h];
  }
  sd[n * HH + h] = s1 / np;
  sa[n * HH + h] = s2 / (np * np);
}

// ---------------- K4: build weights W_b[n,h,u]; pack W0,W1 as f16 B-fragments ----------------
__global__ void k_weights(const float* __restrict__ c00, const float* __restrict__ c01,
                          const float* __restrict__ c10, const float* __restrict__ c11,
                          const float* __restrict__ alpha0, const float* __restrict__ npart,
                          float* __restrict__ wops, half_t* __restrict__ wh0,
                          half_t* __restrict__ wh1) {
  int n = blockIdx.x, h = threadIdx.x;     // 64 threads = h
  float ratio = npart[n] / AVGP;
  float cb[15];
  #pragma unroll
  for (int b = 0; b < 15; ++b) {
    float m = 1.0f;
    if (b >= 5) m = powf(ratio, alpha0[h * 10 + (b - 5)]);
    cb[b] = c00[h * 15 + b] * m;
  }
  const int S = NN * HH * UU;              // slab stride for wops[b-2]
  for (int u = 0; u < UU; ++u) {
    float t = c10[h * UU + u] * c11[h * UU + u];
    float w0 = cb[0] * c01[0 * UU + u] * t;
    float w1 = cb[1] * c01[1 * UU + u] * t;
    #pragma unroll
    for (int b = 2; b < 15; ++b)
      wops[(b - 2) * S + ((n * HH + h) * UU + u)] = cb[b] * c01[b * UU + u] * t;
    // WMMA B-fragment packing: k=h, for u-tile w, k-chunk kc, lane L, elem e:
    //   k = e + 16*(L>>4) + 32*kc ; u = 16*w + (L&15)
    int k = h;
    int e   = k & 15;
    int lhi = (k >> 4) & 1;
    int kc  = k >> 5;
    int L   = (u & 15) | (lhi << 4);
    int w   = u >> 4;
    int idx = n * 4096 + ((w * 2 + kc) * 32 + L) * 16 + e;
    wh0[idx] = (half_t)w0;
    wh1[idx] = (half_t)w1;
  }
}

// ---------------- K5: Cvec (j-dep), Dvec (i-dep, + const terms + bias), Fvec (diag) ----------------
__global__ void k_vectors(const float* __restrict__ wops, const float* __restrict__ dg,
                          const float* __restrict__ sc, const float* __restrict__ sr,
                          const float* __restrict__ sd, const float* __restrict__ sa,
                          const float* __restrict__ npart,
                          const float* __restrict__ bias, const float* __restrict__ dbias,
                          float* __restrict__ cvec, float* __restrict__ dvec,
                          float* __restrict__ fvec) {
  int nb = blockIdx.x;                     // n*128 + p
  int n = nb >> 7;
  int u = threadIdx.x;                     // 0..63
  float inp = 1.0f / npart[n];
  const int S = NN * HH * UU;
  float ac = 0.f, ad = 0.f, af = 0.f;
  for (int h = 0; h < HH; ++h) {
    float d   = dg[nb * HH + h];
    float scp = sc[nb * HH + h] * inp;
    float srp = sr[nb * HH + h] * inp;
    float sdv = sd[n * HH + h];
    float sav = sa[n * HH + h];
    const float* W = wops + (n * HH + h) * UU + u;
    // slabs (b-2): b=2->0, 3->1, 4->2, 5->3, 6->4, 7->5, 8->6, 9->7, 10->8, 11->9, 12->10, 13->11, 14->12
    ac += W[0 * S] * d + W[3 * S] * scp + W[4 * S] * srp;                       // ops 2,5,6 (j-dep)
    ad += W[1 * S] * d + W[6 * S] * scp + W[7 * S] * srp                        // ops 3,8,9 (i-dep)
        + W[5 * S] * sdv + W[11 * S] * sav;                                     // ops 7,13 (const)
    af += W[2 * S] * d + W[8 * S] * sdv + W[9 * S] * srp                        // ops 4,10,11
        + W[10 * S] * scp + W[12 * S] * sav;                                    // ops 12,14
  }
  cvec[nb * UU + u] = ac;
  dvec[nb * UU + u] = ad + bias[u];
  fvec[nb * UU + u] = af + dbias[u];
}

// ---------------- K6: main WMMA kernel ----------------
// Block = 128 threads (4 waves); output tile = (n, i fixed, j in [j0,j0+16), all 64 u).
// Wave w handles u in [16w, 16w+16). acc = A0*B0 + A1*B1 over K=64 (2 chunks of 32).
__global__ __launch_bounds__(128) void k_main(const float* __restrict__ x,
    const unsigned char* __restrict__ mask,
    const half_t* __restrict__ wh0, const half_t* __restrict__ wh1,
    const float* __restrict__ cvec, const float* __restrict__ dvec,
    const float* __restrict__ fvec, float* __restrict__ out) {
  __shared__ half_t sA0[16 * HH];          // x[n,i,j0+m,:]  (rows m, cols k=h)
  __shared__ half_t sA1[16 * HH];          // x[n,j0+m,i,:]

  int bx = blockIdx.x;
  int jt = bx & 7;
  int i  = (bx >> 3) & (PP - 1);
  int n  = bx >> 10;
  int j0 = jt * 16;
  int t  = threadIdx.x;

  // --- stage both A tiles to LDS as f16 ---
  {
    int rr = t >> 3;
    int hb = (t & 7) * 8;
    const float* p0 = x + (((size_t)n * PP + i) * PP + (j0 + rr)) * HH + hb;
    const float* p1 = x + (((size_t)n * PP + (j0 + rr)) * PP + i) * HH + hb;
    float4 a0 = *(const float4*)(p0);
    float4 a1 = *(const float4*)(p0 + 4);
    float4 b0 = *(const float4*)(p1);
    float4 b1 = *(const float4*)(p1 + 4);
    half_t* q0 = &sA0[rr * HH + hb];
    half_t* q1 = &sA1[rr * HH + hb];
    q0[0] = (half_t)a0.x; q0[1] = (half_t)a0.y; q0[2] = (half_t)a0.z; q0[3] = (half_t)a0.w;
    q0[4] = (half_t)a1.x; q0[5] = (half_t)a1.y; q0[6] = (half_t)a1.z; q0[7] = (half_t)a1.w;
    q1[0] = (half_t)b0.x; q1[1] = (half_t)b0.y; q1[2] = (half_t)b0.z; q1[3] = (half_t)b0.w;
    q1[4] = (half_t)b1.x; q1[5] = (half_t)b1.y; q1[6] = (half_t)b1.z; q1[7] = (half_t)b1.w;
  }
  __syncthreads();

  int wv   = t >> 5;
  int L    = t & 31;
  int mrow = L & 15;
  int lhi  = L >> 4;

  // A fragments (contiguous k order, matching B packing: k = e + 16*(L>>4) [+32*kc])
  const half_t* a0base = &sA0[mrow * HH + lhi * 16];
  const half_t* a1base = &sA1[mrow * HH + lhi * 16];
  v16h A00 = load_frag(a0base,      a0base + 8);
  v16h A01 = load_frag(a0base + 32, a0base + 40);
  v16h A10 = load_frag(a1base,      a1base + 8);
  v16h A11 = load_frag(a1base + 32, a1base + 40);

  // B fragments (pre-swizzled by k_weights: contiguous 16 halves per lane)
  const half_t* b0base = wh0 + (size_t)n * 4096 + ((size_t)wv * 2 * 32 + L) * 16;
  const half_t* b1base = wh1 + (size_t)n * 4096 + ((size_t)wv * 2 * 32 + L) * 16;
  v16h B00 = load_frag(b0base,       b0base + 8);
  v16h B01 = load_frag(b0base + 512, b0base + 520);
  v16h B10 = load_frag(b1base,       b1base + 8);
  v16h B11 = load_frag(b1base + 512, b1base + 520);

  v8f acc = {};
  acc = __builtin_amdgcn_wmma_f32_16x16x32_f16(false, A00, false, B00, (short)0, acc, false, false);
  acc = __builtin_amdgcn_wmma_f32_16x16x32_f16(false, A01, false, B01, (short)0, acc, false, false);
  acc = __builtin_amdgcn_wmma_f32_16x16x32_f16(false, A10, false, B10, (short)0, acc, false, false);
  acc = __builtin_amdgcn_wmma_f32_16x16x32_f16(false, A11, false, B11, (short)0, acc, false, false);

  // --- epilogue: C/D layout lane L, VGPR r -> M = r + 8*(L>>4), N = L&15 ---
  int u = wv * 16 + (L & 15);
  float dterm = dvec[(n * PP + i) * UU + u];
  float fterm = fvec[(n * PP + i) * UU + u];
  #pragma unroll
  for (int r = 0; r < 8; ++r) {
    int j = j0 + r + lhi * 8;
    float v = acc[r] + dterm + cvec[(n * PP + j) * UU + u];
    if (i == j) v += fterm;
    v = mask[(size_t)(n * PP + i) * PP + j] ? v : 0.0f;
    out[(((size_t)n * PP + i) * PP + j) * UU + u] = v;
  }
}

extern "C" void kernel_launch(void* const* d_in, const int* in_sizes, int n_in,
                              void* d_out, int out_size, void* d_ws, size_t ws_size,
                              hipStream_t stream) {
  (void)in_sizes; (void)n_in; (void)out_size; (void)ws_size;
  const float* x       = (const float*)d_in[0];
  const unsigned char* mask = (const unsigned char*)d_in[1];
  const float* npart   = (const float*)d_in[2];
  const float* alpha0  = (const float*)d_in[3];
  const float* c00     = (const float*)d_in[4];
  const float* c01     = (const float*)d_in[5];
  const float* c10     = (const float*)d_in[6];
  const float* c11     = (const float*)d_in[7];
  const float* bias    = (const float*)d_in[8];
  const float* dbias   = (const float*)d_in[9];
  float* out = (float*)d_out;

  char* ws = (char*)d_ws;
  float*  sr   = (float*)(ws + 0);         // N*P*H f32  = 128KB
  float*  sc   = (float*)(ws + 131072);    // 128KB
  float*  dg   = (float*)(ws + 262144);    // 128KB
  float*  sd   = (float*)(ws + 393216);    // 1KB (padded)
  float*  sa   = (float*)(ws + 397312);    // 1KB (padded)
  float*  wops = (float*)(ws + 401408);    // 13 * N*H*U f32 = 832KB
  half_t* wh0  = (half_t*)(ws + 1253376);  // N*4096 f16 = 32KB (B-fragment packed)
  half_t* wh1  = (half_t*)(ws + 1286144);  // 32KB
  float*  cvec = (float*)(ws + 1318912);   // N*P*U = 128KB
  float*  dvec = (float*)(ws + 1449984);   // 128KB
  float*  fvec = (float*)(ws + 1581056);   // 128KB  (total ~1.7MB)

  k_rowsum_diag<<<NN * PP, HH, 0, stream>>>(x, sr, dg);
  k_colsum     <<<NN * PP, HH, 0, stream>>>(x, sc);
  k_scalars    <<<NN, HH, 0, stream>>>(sr, dg, npart, sd, sa);
  k_weights    <<<NN, HH, 0, stream>>>(c00, c01, c10, c11, alpha0, npart, wops, wh0, wh1);
  k_vectors    <<<NN * PP, UU, 0, stream>>>(wops, dg, sc, sr, sd, sa, npart, bias, dbias,
                                            cvec, dvec, fvec);
  k_main       <<<NN * PP * (PP / 16), 128, 0, stream>>>(x, mask, wh0, wh1, cvec, dvec, fvec, out);
}
